// CholeskyPlusConst_18674517803682
// MI455X (gfx1250) — matrix operand order repaired and checked by reading, not compiled, not run on hardware
//
#include <hip/hip_runtime.h>

#define NDIM   12
#define TRI    78
#define HID    512
#define BATCH  262144

typedef _Float16 v16h  __attribute__((ext_vector_type(16)));
typedef float    v8f   __attribute__((ext_vector_type(8)));
typedef float    f32x4 __attribute__((ext_vector_type(4)));
typedef unsigned u32x4 __attribute__((ext_vector_type(4)));
typedef _Float16 h2v   __attribute__((ext_vector_type(2)));
typedef int      gi32x4 __attribute__((vector_size(16)));   // matches builtin param

#if defined(__AMDGCN__) && __has_builtin(__builtin_amdgcn_global_load_async_to_lds_b128)
#define HAVE_ASYNC 1
#else
#define HAVE_ASYNC 0
#endif

#if HAVE_ASYNC
#if __has_builtin(__builtin_amdgcn_s_wait_asynccnt)
#define ASYNC_WAIT(n) __builtin_amdgcn_s_wait_asynccnt(n)
#else
#define ASYNC_WAIT(n) asm volatile("s_wait_asynccnt %0" ::"i"(n) : "memory")
#endif
#endif

union U16 { u32x4 q[2]; unsigned u[8]; v16h h; };

static __device__ __forceinline__ unsigned pkrtz(float a, float b) {
    auto t = __builtin_amdgcn_cvt_pkrtz(a, b);     // v_cvt_pk_f16_f32 (RTZ)
    return __builtin_bit_cast(unsigned, t);
}
static __device__ __forceinline__ unsigned pk_rne(float a, float b) {
    h2v t; t.x = (_Float16)a; t.y = (_Float16)b;
    return __builtin_bit_cast(unsigned, t);
}

// ---------------------------------------------------------------------------
// Weight conversion: fp32 -> fp16 in WMMA A-fragment layout (A = W^T tiles).
//   lane<16 : M=lane,    K = {0..7, 16..23}  (dword d -> K pair 2*(d&3)+16*(d>>2))
//   lane>=16: M=lane-16, K = {8..15, 24..31} (+8)
// Regions (dwords): W1A 8192 | W2A 131072 | W3A 20480   (total 624 KB in d_ws)
// ---------------------------------------------------------------------------
__global__ void convert_weights(const float* __restrict__ W1,
                                const float* __restrict__ W2,
                                const float* __restrict__ W3,
                                unsigned* __restrict__ W1A,
                                unsigned* __restrict__ W2A,
                                unsigned* __restrict__ W3A) {
    const int t = blockIdx.x * 256 + threadIdx.x;
    const int N1 = 32 * 32 * 8;           // 8192
    const int N2 = 16 * 32 * 32 * 8;      // 131072
    const int N3 = 16 * 5 * 32 * 8;       // 20480
    if (t < N1) {
        int d = t & 7, lane = (t >> 3) & 31, mt = (t >> 8) & 31;
        int hi = lane >> 4, n = mt * 16 + (lane & 15);
        int ko = 2 * (d & 3) + 16 * (d >> 2) + 8 * hi;
        float a = (ko     < NDIM) ? W1[ko * HID + n]       : 0.f;
        float b = (ko + 1 < NDIM) ? W1[(ko + 1) * HID + n] : 0.f;
        W1A[t] = pk_rne(a, b);
    } else if (t < N1 + N2) {
        int u = t - N1;
        int d = u & 7, lane = (u >> 3) & 31, mt = (u >> 8) & 31, kt = u >> 13;
        int hi = lane >> 4, n = mt * 16 + (lane & 15);
        int k = kt * 32 + 2 * (d & 3) + 16 * (d >> 2) + 8 * hi;
        W2A[u] = pk_rne(W2[k * HID + n], W2[(k + 1) * HID + n]);
    } else if (t < N1 + N2 + N3) {
        int u = t - N1 - N2;
        int d = u & 7, lane = (u >> 3) & 31, f = u >> 8;
        int kt = f / 5, mt = f % 5;
        int hi = lane >> 4, n = mt * 16 + (lane & 15);
        int k = kt * 32 + 2 * (d & 3) + 16 * (d >> 2) + 8 * hi;
        float a = (n < TRI + 1) ? W3[k * (TRI + 1) + n]       : 0.f;
        float b = (n < TRI + 1) ? W3[(k + 1) * (TRI + 1) + n] : 0.f;
        W3A[u] = pk_rne(a, b);
    }
}

template<int I>
static __device__ __forceinline__ void mmt_row(const float (&fl)[80], float* dst) {
    float row[12];
#pragma unroll
    for (int j = 0; j < 12; ++j) {
        const int m = (I < j) ? I : j;
        float s = 0.f;
#pragma unroll
        for (int tt = 0; tt <= m; ++tt)
            s = fmaf(fl[I * (I + 1) / 2 + tt], fl[j * (j + 1) / 2 + tt], s);
        row[j] = s;
    }
#pragma unroll
    for (int j4 = 0; j4 < 3; ++j4) {
        f32x4 v = { row[4 * j4], row[4 * j4 + 1], row[4 * j4 + 2], row[4 * j4 + 3] };
        *(f32x4*)(dst + 4 * j4) = v;
    }
}

// ---------------------------------------------------------------------------
// Fused MLP + softplus + L*L^T. Block = 256 (8 waves), 16 batch rows / wave.
// Dynamic LDS: [0,24KB) W2 slice ring (3 x 8KB), then 16KB h2 frag per wave.
// W2 slices (8 KB = one kt x 8 mt-tile group) are software-pipelined:
//   async path : GLOBAL_LOAD_ASYNC_TO_LDS_B128, triple-buffered, ASYNCcnt waits
//   fallback   : global->VGPR->ds_store, double-buffered, 1 barrier per slice
// ---------------------------------------------------------------------------
__global__ __launch_bounds__(256) void fused_mlp_chol(
        const float* __restrict__ x,
        const unsigned* __restrict__ W1A,
        const unsigned* __restrict__ W2A,
        const unsigned* __restrict__ W3A,
        const float* __restrict__ b1,
        const float* __restrict__ b2,
        const float* __restrict__ b3,
        float* __restrict__ out) {
    extern __shared__ char smem[];
    unsigned* ldsW = (unsigned*)smem;                                   // 24 KB ring
    const int wave = threadIdx.x >> 5;
    const int lane = threadIdx.x & 31;
    const int hi   = lane >> 4;
    const int col  = lane & 15;
    unsigned* h2b = (unsigned*)(smem + 24576 + wave * 16384);           // 16 KB/wave
    const long b0 = (long)blockIdx.x * 128 + wave * 16;
    const long b  = b0 + col;

    // ---- x -> B fragment (x^T, K padded 12->32) -------------------------
    U16 xb;
#pragma unroll
    for (int d = 0; d < 8; ++d) xb.u[d] = 0u;
    if (!hi) {
        const f32x4* xr = (const f32x4*)(x + b * NDIM);
        f32x4 x0 = xr[0], x1 = xr[1];
        xb.u[0] = pkrtz(x0.x, x0.y); xb.u[1] = pkrtz(x0.z, x0.w);
        xb.u[2] = pkrtz(x1.x, x1.y); xb.u[3] = pkrtz(x1.z, x1.w);
    } else {
        const f32x4* xr = (const f32x4*)(x + b * NDIM);
        f32x4 x2 = xr[2];
        xb.u[0] = pkrtz(x2.x, x2.y); xb.u[1] = pkrtz(x2.z, x2.w);
    }

    // ---- Layer 1: h1 = relu(W1^T x x^T), kept in registers --------------
    v16h h1b[16];
    unsigned psave[4];
#pragma unroll
    for (int mt = 0; mt < 32; ++mt) {
        U16 a;
        const u32x4* ap = (const u32x4*)(W1A + (mt * 32 + lane) * 8);
        a.q[0] = ap[0]; a.q[1] = ap[1];
        v8f c = {};
        c = __builtin_amdgcn_wmma_f32_16x16x32_f16(false, a.h, false, xb.h,
                                                   (short)0, c, false, false);
        const float* bp = b1 + mt * 16 + hi * 8;
        unsigned p[4];
#pragma unroll
        for (int d = 0; d < 4; ++d) {
            float va = fmaxf(c[2 * d]     + bp[2 * d],     0.f);
            float vb = fmaxf(c[2 * d + 1] + bp[2 * d + 1], 0.f);
            p[d] = pkrtz(va, vb);
        }
        if ((mt & 1) == 0) {
#pragma unroll
            for (int d = 0; d < 4; ++d) psave[d] = p[d];
        } else {
            U16 f;
#pragma unroll
            for (int d = 0; d < 4; ++d) {
                unsigned q0 = __shfl_xor(psave[d], 16);
                unsigned q1 = __shfl_xor(p[d], 16);
                f.u[d]     = psave[d];
                f.u[4 + d] = p[d];
            }
            h1b[mt >> 1] = f.h;
        }
    }

    // ---- Layer 2: h2 = relu(W2^T x h1), pipelined LDS staging -----------
    auto slice_src = [&](int s) -> const unsigned* {
        // slice s = (g = s>>4, kt = s&15): fragments (kt, g*8 .. g*8+7), 2048 dw
        return W2A + (size_t)(((s & 15) * 32) + ((s >> 4) * 8)) * 256 + threadIdx.x * 8;
    };

#if HAVE_ASYNC
    auto issue_slice = [&](int s, int bufi) {
        gi32x4* gp = (gi32x4*)const_cast<unsigned*>(slice_src(s));
        gi32x4* lp = (gi32x4*)(ldsW + bufi * 2048 + threadIdx.x * 8);
        __builtin_amdgcn_global_load_async_to_lds_b128(gp,     lp,     0, 0);
        __builtin_amdgcn_global_load_async_to_lds_b128(gp + 1, lp + 1, 0, 0);
    };
    issue_slice(0, 0);
    issue_slice(1, 1);
#else
    u32x4 rc0, rc1;
    { const u32x4* sp = (const u32x4*)slice_src(0); rc0 = sp[0]; rc1 = sp[1]; }
#endif

#pragma unroll 1
    for (int g = 0; g < 4; ++g) {
        v8f acc[8];
#pragma unroll
        for (int m = 0; m < 8; ++m) acc[m] = (v8f){};
#pragma unroll
        for (int kt = 0; kt < 16; ++kt) {
            const int s = g * 16 + kt;
            const unsigned* cbuf;
#if HAVE_ASYNC
            if (s + 2 < 64) { issue_slice(s + 2, (s + 2) % 3); ASYNC_WAIT(4); }
            else if (s + 1 < 64) { ASYNC_WAIT(2); }
            else { ASYNC_WAIT(0); }
            __syncthreads();
            cbuf = ldsW + (s % 3) * 2048;
#else
            u32x4 rn0 = rc0, rn1 = rc1;
            if (s + 1 < 64) { const u32x4* sp = (const u32x4*)slice_src(s + 1); rn0 = sp[0]; rn1 = sp[1]; }
            {   u32x4* dp = (u32x4*)(ldsW + (s & 1) * 2048) + threadIdx.x * 2;
                dp[0] = rc0; dp[1] = rc1; }
            __syncthreads();
            cbuf = ldsW + (s & 1) * 2048;
            rc0 = rn0; rc1 = rn1;
#endif
#pragma unroll
            for (int m = 0; m < 8; ++m) {
                U16 a;
                const u32x4* ap = (const u32x4*)(cbuf + (m * 32 + lane) * 8);
                a.q[0] = ap[0]; a.q[1] = ap[1];
                acc[m] = __builtin_amdgcn_wmma_f32_16x16x32_f16(false, a.h, false, h1b[kt],
                                                                (short)0, acc[m], false, false);
            }
        }
        // epilogue: bias + relu + pack -> h2 B-frags in LDS
#pragma unroll
        for (int m = 0; m < 8; ++m) {
            const float* bp = b2 + (g * 8 + m) * 16 + hi * 8;
            unsigned p[4];
#pragma unroll
            for (int d = 0; d < 4; ++d) {
                float va = fmaxf(acc[m][2 * d]     + bp[2 * d],     0.f);
                float vb = fmaxf(acc[m][2 * d + 1] + bp[2 * d + 1], 0.f);
                p[d] = pkrtz(va, vb);
            }
            if ((m & 1) == 0) {
#pragma unroll
                for (int d = 0; d < 4; ++d) psave[d] = p[d];
            } else {
                U16 f;
#pragma unroll
                for (int d = 0; d < 4; ++d) {
                    unsigned q0 = __shfl_xor(psave[d], 16);
                    unsigned q1 = __shfl_xor(p[d], 16);
                    f.u[d]     = psave[d];
                    f.u[4 + d] = p[d];
                }
                int kt2 = g * 4 + (m >> 1);
                u32x4* dp = (u32x4*)(h2b + (kt2 * 32 + lane) * 8);
                dp[0] = f.q[0]; dp[1] = f.q[1];
            }
        }
    }

    // ---- Layer 3: out = W3^T x h2 (5 output tiles, K=512) ---------------
    v8f acc3[5];
#pragma unroll
    for (int m = 0; m < 5; ++m) acc3[m] = (v8f){};
#pragma unroll 1
    for (int kt = 0; kt < 16; ++kt) {
        U16 bf;
        const u32x4* hp = (const u32x4*)(h2b + (kt * 32 + lane) * 8);
        bf.q[0] = hp[0]; bf.q[1] = hp[1];
#pragma unroll
        for (int mt = 0; mt < 5; ++mt) {
            U16 a;
            const u32x4* ap = (const u32x4*)(W3A + ((size_t)(kt * 5 + mt) * 32 + lane) * 8);
            a.q[0] = ap[0]; a.q[1] = ap[1];
            acc3[mt] = __builtin_amdgcn_wmma_f32_16x16x32_f16(false, a.h, false, bf.h,
                                                              (short)0, acc3[mt], false, false);
        }
    }

    // ---- softplus + gather full 79-vector per batch row ------------------
    float full[80];
#pragma unroll
    for (int mt = 0; mt < 5; ++mt) {
#pragma unroll
        for (int r = 0; r < 8; ++r) {
            int n = mt * 16 + hi * 8 + r;
            float bias = b3[n <= TRI ? n : TRI];
            float v = acc3[mt][r] + bias;
            float sp = fmaxf(v, 0.f) + __logf(1.f + __expf(-fabsf(v)));
            float other = __shfl_xor(sp, 16);
            full[mt * 16 + r]     = hi ? other : sp;
            full[mt * 16 + 8 + r] = hi ? sp    : other;
        }
    }

    // ---- MMT = L * L^T, balanced split across lane halves ----------------
    float* po = out + (size_t)b * 144;
    if (!hi) {
        mmt_row<0>(full, po + 0 * 12);
        mmt_row<2>(full, po + 2 * 12);
        mmt_row<4>(full, po + 4 * 12);
        mmt_row<7>(full, po + 7 * 12);
        mmt_row<9>(full, po + 9 * 12);
        mmt_row<11>(full, po + 11 * 12);
        out[(size_t)BATCH * 144 + b] = full[TRI];     // c channel
    } else {
        mmt_row<1>(full, po + 1 * 12);
        mmt_row<3>(full, po + 3 * 12);
        mmt_row<5>(full, po + 5 * 12);
        mmt_row<6>(full, po + 6 * 12);
        mmt_row<8>(full, po + 8 * 12);
        mmt_row<10>(full, po + 10 * 12);
    }
}

extern "C" void kernel_launch(void* const* d_in, const int* in_sizes, int n_in,
                              void* d_out, int out_size, void* d_ws, size_t ws_size,
                              hipStream_t stream) {
    const float* x  = (const float*)d_in[0];
    const float* W1 = (const float*)d_in[1];
    const float* b1 = (const float*)d_in[2];
    const float* W2 = (const float*)d_in[3];
    const float* b2 = (const float*)d_in[4];
    const float* W3 = (const float*)d_in[5];
    const float* b3 = (const float*)d_in[6];

    unsigned* W1A = (unsigned*)d_ws;       // 32 KB
    unsigned* W2A = W1A + 8192;            // 512 KB
    unsigned* W3A = W2A + 131072;          // 80 KB

    convert_weights<<<624, 256, 0, stream>>>(W1, W2, W3, W1A, W2A, W3A);

    const size_t smem = 24576 + 8 * 16384; // 152 KB dynamic LDS (WGP has 320 KB)
    (void)hipFuncSetAttribute(reinterpret_cast<const void*>(&fused_mlp_chol),
                              hipFuncAttributeMaxDynamicSharedMemorySize, (int)smem);
    fused_mlp_chol<<<BATCH / 128, 256, smem, stream>>>(x, W1A, W2A, W3A, b1, b2, b3,
                                                       (float*)d_out);
}
